// CongressGAT_10118942949378
// MI455X (gfx1250) — matrix operand order repaired and checked by reading, not compiled, not run on hardware
//
#include <hip/hip_runtime.h>

#define GN   4096   // nodes
#define GH   4      // heads
#define GF   32     // features per head
#define GKIN 128    // input features (both layers: 128 and H*F=128)

typedef __bf16 v16bf __attribute__((ext_vector_type(16)));
typedef float  v8f   __attribute__((ext_vector_type(8)));
typedef float  v4f   __attribute__((ext_vector_type(4)));   // native vector for NT stores

// ---------------------------------------------------------------------------
// Kernel 1: per-node feature transform  h[hd][n][f] = sum_k x[n][k]*W[hd][k][f]
// plus attention coefficients s_src/s_dst via wave32 reductions.
// Block = 128 threads = 4 waves; wave hd handles head hd, lane = f.
// ---------------------------------------------------------------------------
__global__ void gat_feat(const float* __restrict__ x,
                         const float* __restrict__ W,
                         const float* __restrict__ asrc,
                         const float* __restrict__ adst,
                         float* __restrict__ hfeat,   // [H][N][F]
                         float* __restrict__ ssrc,    // [H][N]
                         float* __restrict__ sdst)    // [H][N]
{
    __shared__ float xs[GKIN];
    const int n   = blockIdx.x;
    const int tid = threadIdx.x;           // 0..127
    xs[tid] = x[(size_t)n * GKIN + tid];
    __syncthreads();

    const int hd = tid >> 5;
    const int f  = tid & 31;
    const float* w = W + (size_t)hd * GKIN * GF + f;
    float acc = 0.f;
#pragma unroll 8
    for (int k = 0; k < GKIN; ++k)
        acc = fmaf(xs[k], w[(size_t)k * GF], acc);

    hfeat[((size_t)hd * GN + n) * GF + f] = acc;

    float ps = acc * asrc[hd * GF + f];
    float pd = acc * adst[hd * GF + f];
#pragma unroll
    for (int o = 16; o > 0; o >>= 1) {
        ps += __shfl_xor(ps, o, 32);
        pd += __shfl_xor(pd, o, 32);
    }
    if (f == 0) {
        ssrc[hd * GN + n] = ps;
        sdst[hd * GN + n] = pd;
    }
}

// ---------------------------------------------------------------------------
// Kernel 2: pre-swizzle h into CDNA5 WMMA B-matrix (32x16, 16-bit) lane layout
// as bf16.  bswz[hd][jt][ftile][lane][e]  with  K = (e>>3)*16 + (lane>>4)*8 + (e&7),
// N-col = ftile*16 + (lane&15).  Hot loop then loads 32B/lane contiguously.
// ---------------------------------------------------------------------------
__global__ void gat_swz(const float* __restrict__ hfeat, __bf16* __restrict__ bswz)
{
    const int idx  = blockIdx.x * blockDim.x + threadIdx.x; // < H*(N/32)*2*32*16
    const int e    = idx & 15;
    const int lane = (idx >> 4) & 31;
    const int ft   = (idx >> 9) & 1;
    const int jt   = (idx >> 10) & (GN / 32 - 1);
    const int hd   = idx >> 17;
    const int jrow = jt * 32 + ((e >> 3) << 4) + (lane >> 4) * 8 + (e & 7);
    const int f    = ft * 16 + (lane & 15);
    bswz[idx] = (__bf16)hfeat[((size_t)hd * GN + jrow) * GF + f];
}

// ---------------------------------------------------------------------------
// Kernel 3: softmax statistics (masked row max + sum of exp), branch-free.
// adj values are exactly {0,1}: mask logits with one FMA instead of branches.
// Empty rows keep the -3e38 sentinel max (their rowsum is bogus but unused:
// pass2 multiplies every element by adj=0).
// One wave per (row, head); block = 256 = 8 rows. adj is L2-resident (64MB).
// ---------------------------------------------------------------------------
__global__ void gat_pass1(const float* __restrict__ adj,
                          const float* __restrict__ ssrc,
                          const float* __restrict__ sdst,
                          float* __restrict__ rowm,
                          float* __restrict__ rowsum)
{
    const int hd   = blockIdx.y;
    const int row  = blockIdx.x * 8 + (threadIdx.x >> 5);
    const int lane = threadIdx.x & 31;
    const float* adjrow = adj + (size_t)row * GN;
    const float* sd     = sdst + hd * GN;
    const float  ss     = ssrc[hd * GN + row];
    const float  BIG    = 3.0e38f;

    float m = -BIG;
    for (int j = lane * 4; j < GN; j += 128) {
        v4f a4 = *(const v4f*)(adjrow + j);
        v4f d4 = *(const v4f*)(sd + j);
#pragma unroll
        for (int q = 0; q < 4; ++q) {
            float l = ss + d4[q];
            l = l >= 0.f ? l : 0.2f * l;          // LeakyReLU(0.2)
            l = fmaf(a4[q], l + BIG, -BIG);       // adj=1 -> l ; adj=0 -> -BIG
            m = fmaxf(m, l);
        }
    }
#pragma unroll
    for (int o = 16; o > 0; o >>= 1) m = fmaxf(m, __shfl_xor(m, o, 32));

    float s = 0.f;
    for (int j = lane * 4; j < GN; j += 128) {
        v4f a4 = *(const v4f*)(adjrow + j);
        v4f d4 = *(const v4f*)(sd + j);
#pragma unroll
        for (int q = 0; q < 4; ++q) {
            float l = ss + d4[q];
            l = l >= 0.f ? l : 0.2f * l;
            l = fmaf(a4[q], l + BIG, -BIG);       // masked -> exp underflows to 0
            s += __expf(fminf(l - m, 60.f));
        }
    }
#pragma unroll
    for (int o = 16; o > 0; o >>= 1) s += __shfl_xor(s, o, 32);

    if (lane == 0) { rowm[hd * GN + row] = m; rowsum[hd * GN + row] = s; }
}

// ---------------------------------------------------------------------------
// Kernel 4 (hot): block = 128 threads = 4 waves, one 16-row tile per block.
// Wave w handles column-tiles jt = w, w+4, ... with private WMMA accumulators;
// 8KB LDS reduction at the end.  Per jt:
//  - build 16x32 attention tile directly in the WMMA 16-bit A lane layout
//    (lane: r = lane&15, kh = lane>>4; e<8 -> K=kh*8+e, e>=8 -> K=16+kh*8+e-8)
//  - branch-free masked softmax value:  exp(min(l-m,60)) * inv * adj
//  - NON-TEMPORAL fp32 attention stores (write-once 537MB stream; keep the
//    L2-resident adj working set from being evicted)
//  - two v_wmma_f32_16x16x32_bf16 against pre-swizzled bf16 h (F=32 -> 2 tiles)
// ---------------------------------------------------------------------------
__global__ void gat_pass2(const float* __restrict__ adj,
                          const float* __restrict__ ssrc,
                          const float* __restrict__ sdst,
                          const float* __restrict__ rowm,
                          const float* __restrict__ rowsum,
                          const __bf16* __restrict__ bswz,
                          float* __restrict__ attn_out,   // [H][N][N]
                          float* __restrict__ outh)       // [H][N][F]
{
    __shared__ float red[4][16][32];   // wave, vreg (c0:0-7, c1:8-15), lane

    const int hd   = blockIdx.y;
    const int i0   = blockIdx.x * 16;
    const int wave = threadIdx.x >> 5;
    const int lane = threadIdx.x & 31;
    const int r    = lane & 15;
    const int kh   = lane >> 4;
    const int row  = i0 + r;

    const float ss  = ssrc[hd * GN + row];
    const float m   = rowm[hd * GN + row];
    const float sm  = rowsum[hd * GN + row];
    const float inv = sm > 0.f ? 1.f / sm : 0.f;

    const float* adjrow  = adj + (size_t)row * GN;
    const float* sd      = sdst + hd * GN;
    float*       attnrow = attn_out + ((size_t)hd * GN + row) * GN;
    const __bf16* bbase  = bswz + (size_t)hd * (GN / 32) * 2 * 32 * 16;

    v8f c0 = {};
    v8f c1 = {};

    for (int jt = wave; jt < GN / 32; jt += 4) {
        const int j  = jt * 32;
        const int b0 = j + kh * 8;        // columns for e = 0..7  (contiguous)
        const int b1 = j + 16 + kh * 8;   // columns for e = 8..15 (contiguous)

        alignas(16) float av[16], dv[16], ov[16];
        *(v4f*)(av + 0)  = *(const v4f*)(adjrow + b0);
        *(v4f*)(av + 4)  = *(const v4f*)(adjrow + b0 + 4);
        *(v4f*)(av + 8)  = *(const v4f*)(adjrow + b1);
        *(v4f*)(av + 12) = *(const v4f*)(adjrow + b1 + 4);
        *(v4f*)(dv + 0)  = *(const v4f*)(sd + b0);
        *(v4f*)(dv + 4)  = *(const v4f*)(sd + b0 + 4);
        *(v4f*)(dv + 8)  = *(const v4f*)(sd + b1);
        *(v4f*)(dv + 12) = *(const v4f*)(sd + b1 + 4);

        v16bf a;
#pragma unroll
        for (int e = 0; e < 16; ++e) {
            float l = ss + dv[e];
            l = l >= 0.f ? l : 0.2f * l;                   // LeakyReLU(0.2)
            // branch-free masked softmax: adj in {0,1}; clamp keeps exp finite
            float v = __expf(fminf(l - m, 60.f)) * inv * av[e];
            ov[e] = v;
            a[e]  = (__bf16)v;
        }

        // mandatory attention output (fp32) -- non-temporal streaming stores
        __builtin_nontemporal_store(*(const v4f*)(ov + 0),  (v4f*)(attnrow + b0));
        __builtin_nontemporal_store(*(const v4f*)(ov + 4),  (v4f*)(attnrow + b0 + 4));
        __builtin_nontemporal_store(*(const v4f*)(ov + 8),  (v4f*)(attnrow + b1));
        __builtin_nontemporal_store(*(const v4f*)(ov + 12), (v4f*)(attnrow + b1 + 4));

        // B fragments: contiguous 32B per lane thanks to pre-swizzle
        const v16bf b0v = *(const v16bf*)(bbase + (((size_t)jt * 2 + 0) * 32 + lane) * 16);
        const v16bf b1v = *(const v16bf*)(bbase + (((size_t)jt * 2 + 1) * 32 + lane) * 16);

        c0 = __builtin_amdgcn_wmma_f32_16x16x32_bf16(false, a, false, b0v,
                                                     (short)0, c0, false, false);
        c1 = __builtin_amdgcn_wmma_f32_16x16x32_bf16(false, a, false, b1v,
                                                     (short)0, c1, false, false);
    }

    // cross-wave reduction of the partial C tiles
#pragma unroll
    for (int v = 0; v < 8; ++v) {
        red[wave][v][lane]     = c0[v];
        red[wave][v + 8][lane] = c1[v];
    }
    __syncthreads();

    if (wave == 0) {
        // C/D layout: VGPR v, lanes 0-15 -> M=v, lanes 16-31 -> M=v+8; N=lane&15
        const int col = lane & 15;
#pragma unroll
        for (int v = 0; v < 8; ++v) {
            const int orow = i0 + v + 8 * kh;
            float s0 = red[0][v][lane] + red[1][v][lane] +
                       red[2][v][lane] + red[3][v][lane];
            float s1 = red[0][v + 8][lane] + red[1][v + 8][lane] +
                       red[2][v + 8][lane] + red[3][v + 8][lane];
            outh[((size_t)hd * GN + orow) * GF + col]      = s0;
            outh[((size_t)hd * GN + orow) * GF + 16 + col] = s1;
        }
    }
}

// ---------------------------------------------------------------------------
// elu(concat heads) -> layer-2 input  x2[n][hd*F+f]
// ---------------------------------------------------------------------------
__global__ void gat_concat(const float* __restrict__ outh, float* __restrict__ x2)
{
    const int idx = blockIdx.x * blockDim.x + threadIdx.x; // < N*H*F
    const int f  = idx & 31;
    const int hd = (idx >> 5) & 3;
    const int n  = idx >> 7;
    float v = outh[((size_t)hd * GN + n) * GF + f];
    v = v > 0.f ? v : (__expf(v) - 1.f);                   // elu
    x2[(size_t)n * (GH * GF) + hd * GF + f] = v;
}

// ---------------------------------------------------------------------------
// elu(mean over heads) -> final h  [N][F]
// ---------------------------------------------------------------------------
__global__ void gat_mean(const float* __restrict__ outh, float* __restrict__ out)
{
    const int idx = blockIdx.x * blockDim.x + threadIdx.x; // < N*F
    const int f = idx & 31;
    const int n = idx >> 5;
    float s = 0.f;
#pragma unroll
    for (int hd = 0; hd < GH; ++hd)
        s += outh[((size_t)hd * GN + n) * GF + f];
    s *= 0.25f;
    s = s > 0.f ? s : (__expf(s) - 1.f);                   // elu
    out[idx] = s;
}

// ---------------------------------------------------------------------------
extern "C" void kernel_launch(void* const* d_in, const int* in_sizes, int n_in,
                              void* d_out, int out_size, void* d_ws, size_t ws_size,
                              hipStream_t stream)
{
    const float* x      = (const float*)d_in[0];
    const float* adj    = (const float*)d_in[1];
    const float* W1     = (const float*)d_in[2];
    const float* asrc1  = (const float*)d_in[3];
    const float* adst1  = (const float*)d_in[4];
    const float* W2     = (const float*)d_in[5];
    const float* asrc2  = (const float*)d_in[6];
    const float* adst2  = (const float*)d_in[7];

    float* out = (float*)d_out;
    float* attn1 = out + (size_t)GN * GF;                    // [H][N][N]
    float* attn2 = attn1 + (size_t)GH * GN * GN;             // [H][N][N]

    // workspace layout (floats); bf16 region kept 32B aligned
    float* ws     = (float*)d_ws;
    float* hfeat  = ws;                                       // H*N*F
    float* outh   = hfeat + (size_t)GH * GN * GF;             // H*N*F
    float* x2     = outh  + (size_t)GH * GN * GF;             // N*KIN
    float* ssrc   = x2    + (size_t)GN * GKIN;                // H*N
    float* sdst   = ssrc  + (size_t)GH * GN;
    float* rowm   = sdst  + (size_t)GH * GN;
    float* rowsum = rowm  + (size_t)GH * GN;
    __bf16* bswz  = (__bf16*)(rowsum + (size_t)GH * GN);      // H*N*F bf16

    const int swzN  = GH * (GN / 32) * 2 * 32 * 16;           // 524288
    const dim3 p1g(GN / 8, GH), p2g(GN / 16, GH);

    // ---- layer 1 ----
    gat_feat  <<<GN, GKIN, 0, stream>>>(x,  W1, asrc1, adst1, hfeat, ssrc, sdst);
    gat_swz   <<<swzN / 256, 256, 0, stream>>>(hfeat, bswz);
    gat_pass1 <<<p1g, 256, 0, stream>>>(adj, ssrc, sdst, rowm, rowsum);
    gat_pass2 <<<p2g, 128, 0, stream>>>(adj, ssrc, sdst, rowm, rowsum, bswz, attn1, outh);
    gat_concat<<<(GN * GH * GF) / 256, 256, 0, stream>>>(outh, x2);

    // ---- layer 2 ----
    gat_feat  <<<GN, GKIN, 0, stream>>>(x2, W2, asrc2, adst2, hfeat, ssrc, sdst);
    gat_swz   <<<swzN / 256, 256, 0, stream>>>(hfeat, bswz);
    gat_pass1 <<<p1g, 256, 0, stream>>>(adj, ssrc, sdst, rowm, rowsum);
    gat_pass2 <<<p2g, 128, 0, stream>>>(adj, ssrc, sdst, rowm, rowsum, bswz, attn2, outh);
    gat_mean  <<<(GN * GF) / 256, 256, 0, stream>>>(outh, out);
}
